// PGCN_39522289058323
// MI455X (gfx1250) — compile-verified, matrix operand order checked by reading, not run on hardware
//
#include <hip/hip_runtime.h>

// ---------------------------------------------------------------------------
// PGCN pipeline for MI455X (gfx1250, wave32, WMMA).
//   embed GEMMs  -> v_wmma_f32_16x16x32_bf16 with the B operand staged through
//                   LDS via global_load_async_to_lds_b128 (double buffered,
//                   s_wait_asynccnt + barrier), A streamed from HBM (mem-bound)
//   score GEMM   -> one bf16 WMMA per 16x16 tile (K=32), output-stream bound
//   graph layers -> global_atomic_add_f32 scatter, float4 gather
// ---------------------------------------------------------------------------

#define N_NODES 8000
#define FEAT_K  8000
#define EDGES   524288      // 2^19
#define KT      250         // 8000 / 32 k-tiles
#define NKCH    5
#define KCHUNK  (KT / NKCH) // 50

typedef __attribute__((ext_vector_type(16))) __bf16 bf16x16;
typedef __attribute__((ext_vector_type(8)))  float  floatx8;

__device__ __forceinline__ unsigned pack_bf16(float lo, float hi) {
  unsigned a = __float_as_uint(lo), b = __float_as_uint(hi);
  a += 0x7FFFu + ((a >> 16) & 1u);          // round-to-nearest-even
  b += 0x7FFFu + ((b >> 16) & 1u);
  return (a >> 16) | (b & 0xFFFF0000u);
}

union Frag { bf16x16 v; unsigned u[8]; uint4 q[2]; };

// Load one WMMA 16x32 bf16 A-fragment (or B-fragment via pre-transposed rows)
// from an f32 row-major row.  ISA 16-bit A layout: lane = 16*khalf + m,
// VGPR0..3 hold K = khalf*8 + 0..7 (contiguous!), VGPR4..7 hold K = 16+khalf*8+0..7.
// => two contiguous 32-byte chunks per lane -> 4x float4 loads.
__device__ __forceinline__ bf16x16 load_frag_f32(const float* __restrict__ row) {
  const int khalf = (threadIdx.x & 31) >> 4;
  const float4* p0 = (const float4*)(row + khalf * 8);
  const float4* p1 = (const float4*)(row + 16 + khalf * 8);
  float4 a0 = p0[0], a1 = p0[1], b0 = p1[0], b1 = p1[1];
  Frag f;
  f.u[0] = pack_bf16(a0.x, a0.y); f.u[1] = pack_bf16(a0.z, a0.w);
  f.u[2] = pack_bf16(a1.x, a1.y); f.u[3] = pack_bf16(a1.z, a1.w);
  f.u[4] = pack_bf16(b0.x, b0.y); f.u[5] = pack_bf16(b0.z, b0.w);
  f.u[6] = pack_bf16(b1.x, b1.y); f.u[7] = pack_bf16(b1.z, b1.w);
  return f.v;
}

// Async copy of one 4 KB swizzled-weight k-tile into LDS: 128 threads x 2 x 16B.
// Tracked on ASYNCcnt; in-order completion lets us double-buffer with
// s_wait_asynccnt <= 2.
__device__ __forceinline__ void async_copy_tile(const unsigned char* __restrict__ src,
                                                unsigned lds_off) {
  const unsigned t16 = threadIdx.x * 16u;
  unsigned long long g0 = (unsigned long long)(uintptr_t)(src + t16);
  unsigned long long g1 = g0 + 2048ull;
  unsigned l0 = lds_off + t16;
  unsigned l1 = l0 + 2048u;
  asm volatile("global_load_async_to_lds_b128 %0, %1, off" :: "v"(l0), "v"(g0) : "memory");
  asm volatile("global_load_async_to_lds_b128 %0, %1, off" :: "v"(l1), "v"(g1) : "memory");
}

// ------------------------------- utilities ---------------------------------
__global__ void k_zero(float* __restrict__ p, int n) {
  int i = blockIdx.x * blockDim.x + threadIdx.x;
  if (i < n) p[i] = 0.0f;
}

// Pre-swizzle W [8000x64] f32 into bf16 B-fragments: for each (ktile, nblock)
// a 1 KB block laid out [lane][8 dwords]; one k-tile = 4 KB contiguous.
__global__ void k_swizzle_w(const float* __restrict__ W, unsigned* __restrict__ ws) {
  int t = blockIdx.x * blockDim.x + threadIdx.x;   // KT*4*32*8 = 256000
  int j = t & 7, lane = (t >> 3) & 31, nb = (t >> 8) & 3, kt = t >> 10;
  int n = lane & 15, khalf = lane >> 4;
  int Klo = (j < 4) ? (khalf * 8 + 2 * j) : (16 + khalf * 8 + 2 * j - 8);
  float lo = W[(size_t)(kt * 32 + Klo)     * 64 + nb * 16 + n];
  float hi = W[(size_t)(kt * 32 + Klo + 1) * 64 + nb * 16 + n];
  ws[t] = pack_bf16(lo, hi);
}

// ------------------------ input embedding GEMM (WMMA) ----------------------
// h[8000x64] = feat[8000x8000] @ W[8000x64].  4 waves / block, one 16-row tile
// each, 4 accumulators covering all 64 output columns.  The shared 4 KB B tile
// for the current k-slice is streamed Global->LDS with async copies (double
// buffered) so all 4 waves read it from LDS and the 4 WMMAs issue back-to-back.
// K split over gridDim.y with atomic f32 finish; bound by the feat HBM stream.
__global__ void k_embed_gemm(const float* __restrict__ feat,
                             const unsigned* __restrict__ wsw,
                             float* __restrict__ h) {
  __shared__ __align__(16) unsigned char ldsB[2][4096];
  const int wave = threadIdx.x >> 5, lane = threadIdx.x & 31;
  const int mt = blockIdx.x * 4 + wave;            // 0..499
  const int kt0 = blockIdx.y * KCHUNK;
  const unsigned lds0 = (unsigned)(uintptr_t)&ldsB[0][0];   // addr[31:0] = LDS offset
  const unsigned char* wbytes = (const unsigned char*)wsw;
  const float* row = feat + (size_t)(mt * 16 + (lane & 15)) * FEAT_K;
  floatx8 acc0 = {}, acc1 = {}, acc2 = {}, acc3 = {};

  async_copy_tile(wbytes + (size_t)kt0 * 4096, lds0);       // prologue: buf 0
  for (int i = 0; i < KCHUNK; ++i) {
    const int kt = kt0 + i;
    const unsigned bufoff = (i & 1) ? 4096u : 0u;
    if (i + 1 < KCHUNK) {
      async_copy_tile(wbytes + (size_t)(kt + 1) * 4096, lds0 + (bufoff ^ 4096u));
      asm volatile("s_wait_asynccnt 0x2" ::: "memory");     // current buf's 2 done
    } else {
      asm volatile("s_wait_asynccnt 0x0" ::: "memory");
    }
    __syncthreads();                                        // all waves' copies visible

    const int k0 = kt * 32;
    __builtin_prefetch(row + k0 + 256, 0, 0);               // global_prefetch_b8
    bf16x16 a = load_frag_f32(row + k0);

    const uint4* bp = (const uint4*)&ldsB[i & 1][lane * 32]; // nb stride = 64 uint4
    Frag b0c, b1c, b2c, b3c;
    b0c.q[0] = bp[0];   b0c.q[1] = bp[1];
    b1c.q[0] = bp[64];  b1c.q[1] = bp[65];
    b2c.q[0] = bp[128]; b2c.q[1] = bp[129];
    b3c.q[0] = bp[192]; b3c.q[1] = bp[193];
    acc0 = __builtin_amdgcn_wmma_f32_16x16x32_bf16(false, a, false, b0c.v, (short)0, acc0, false, false);
    acc1 = __builtin_amdgcn_wmma_f32_16x16x32_bf16(false, a, false, b1c.v, (short)0, acc1, false, false);
    acc2 = __builtin_amdgcn_wmma_f32_16x16x32_bf16(false, a, false, b2c.v, (short)0, acc2, false, false);
    acc3 = __builtin_amdgcn_wmma_f32_16x16x32_bf16(false, a, false, b3c.v, (short)0, acc3, false, false);
    __syncthreads();                                        // reads done before overwrite
  }

  // C/D layout: lane<16 -> M = vgpr, lane>=16 -> M = vgpr+8; N = lane%16
  float* hp = h + (size_t)(mt * 16 + (lane >> 4) * 8) * 64 + (lane & 15);
#pragma unroll
  for (int v = 0; v < 8; ++v) {
    atomicAdd(hp + (size_t)v * 64 +  0, acc0[v]);
    atomicAdd(hp + (size_t)v * 64 + 16, acc1[v]);
    atomicAdd(hp + (size_t)v * 64 + 32, acc2[v]);
    atomicAdd(hp + (size_t)v * 64 + 48, acc3[v]);
  }
}

// ------------------------------ graph layers -------------------------------
__global__ void k_degrees(const int* __restrict__ src, const int* __restrict__ dst,
                          float* __restrict__ ro, float* __restrict__ ri) {
  int gid = blockIdx.x * blockDim.x + threadIdx.x;   // 4*EDGES
  int r = gid >> 19;
  atomicAdd(&ro[r * N_NODES + src[gid]], 1.0f);
  atomicAdd(&ri[r * N_NODES + dst[gid]], 1.0f);
}

__global__ void k_rsqrt_clip(float* __restrict__ p, int n) {
  int i = blockIdx.x * blockDim.x + threadIdx.x;
  if (i < n) p[i] = rsqrtf(fmaxf(p[i], 1.0f));
}

// m[i,:] = (h[i,:] * ro[i]) @ W   (W is 64 x Fout, Fout = 1<<fsh)
__global__ void k_rel_gemm(const float* __restrict__ h, const float* __restrict__ ro,
                           const float* __restrict__ W, float* __restrict__ m, int fsh) {
  int idx = blockIdx.x * blockDim.x + threadIdx.x;   // N_NODES << fsh
  int i = idx >> fsh, j = idx & ((1 << fsh) - 1);
  const float* hr = h + (size_t)i * 64;
  float s = 0.f;
#pragma unroll 8
  for (int k = 0; k < 64; ++k) s += hr[k] * W[(k << fsh) + j];
  m[idx] = s * ro[i];
}

// acc[dst,:] += m[src,:] * rsqrt(deg_in[dst])   — float4 gather, f32 atomics
__global__ void k_scatter(const float* __restrict__ m, const int* __restrict__ src,
                          const int* __restrict__ dst, const float* __restrict__ ri,
                          float* __restrict__ acc, int qsh, int fsh) {
  int gid = blockIdx.x * blockDim.x + threadIdx.x;   // EDGES << qsh
  int e = gid >> qsh;
  int f = (gid & ((1 << qsh) - 1)) << 2;
  int s = src[e], d = dst[e];
  float w = ri[d];
  const float4 v = *(const float4*)(m + ((size_t)s << fsh) + f);
  float* ap = acc + ((size_t)d << fsh) + f;
  atomicAdd(ap + 0, v.x * w);
  atomicAdd(ap + 1, v.y * w);
  atomicAdd(ap + 2, v.z * w);
  atomicAdd(ap + 3, v.w * w);
}

__global__ void k_bias_relu(const float* __restrict__ acc, const float* __restrict__ bA,
                            const float* __restrict__ bB, float* __restrict__ out, int fsh) {
  int idx = blockIdx.x * blockDim.x + threadIdx.x;   // N_NODES << fsh
  int f = idx & ((1 << fsh) - 1);
  out[idx] = fmaxf(acc[idx] + bA[f] + bB[f], 0.0f);
}

// ------------------------------ bilinear score -----------------------------
__global__ void k_bilinear_left(const float* __restrict__ h, const float* __restrict__ Wf,
                                float* __restrict__ P) {
  int idx = blockIdx.x * blockDim.x + threadIdx.x;   // 8000*32
  int i = idx >> 5, j = idx & 31;
  const float* hr = h + (size_t)i * 32;
  const float* wr = Wf + j * 32;                     // P = h @ Wf^T
  float s = 0.f;
#pragma unroll
  for (int k = 0; k < 32; ++k) s += hr[k] * wr[k];
  P[idx] = s;
}

// out[8000x8000] = P[8000x32] @ Hdis[8000x32]^T : K=32 = one bf16 WMMA / tile.
// Memory-bound on the 256 MB output stream; operands are L2-resident.
__global__ void k_score_gemm(const float* __restrict__ P, const float* __restrict__ Hd,
                             float* __restrict__ out) {
  const int wave = threadIdx.x >> 5, lane = threadIdx.x & 31;
  const int mt = blockIdx.x;                 // 0..499
  const int nt = blockIdx.y * 4 + wave;      // 0..499
  bf16x16 a = load_frag_f32(P  + (size_t)(mt * 16 + (lane & 15)) * 32);
  bf16x16 b = load_frag_f32(Hd + (size_t)(nt * 16 + (lane & 15)) * 32);
  floatx8 acc = {};
  acc = __builtin_amdgcn_wmma_f32_16x16x32_bf16(false, a, false, b, (short)0, acc, false, false);
  const int row0 = mt * 16 + (lane >> 4) * 8;
  const int col  = nt * 16 + (lane & 15);
#pragma unroll
  for (int v = 0; v < 8; ++v)
    out[(size_t)(row0 + v) * N_NODES + col] = acc[v];   // 64B coalesced segments
}

// --------------------------------- driver ----------------------------------
extern "C" void kernel_launch(void* const* d_in, const int* in_sizes, int n_in,
                              void* d_out, int out_size, void* d_ws, size_t ws_size,
                              hipStream_t stream) {
  (void)in_sizes; (void)n_in; (void)out_size; (void)ws_size;
  const float* feat_drug = (const float*)d_in[0];
  const float* feat_dis  = (const float*)d_in[1];
  const float* W_drug    = (const float*)d_in[2];
  const float* W_dis     = (const float*)d_in[3];
  const float* W1        = (const float*)d_in[4];
  const float* b1        = (const float*)d_in[5];
  const float* W2        = (const float*)d_in[6];
  const float* b2        = (const float*)d_in[7];
  const float* Wf        = (const float*)d_in[8];
  const int*   src       = (const int*)d_in[9];
  const int*   dst       = (const int*)d_in[10];
  float* out = (float*)d_out;

  char* ws = (char*)d_ws;
  size_t off = 0;
  auto take = [&](size_t bytes) -> void* {
    void* p = ws + off; off += (bytes + 255) & ~(size_t)255; return p;
  };
  unsigned* wbf_drug = (unsigned*)take((size_t)KT * 4 * 32 * 8 * 4);
  unsigned* wbf_dis  = (unsigned*)take((size_t)KT * 4 * 32 * 8 * 4);
  float* h_drug   = (float*)take((size_t)N_NODES * 64 * 4);
  float* h_dis    = (float*)take((size_t)N_NODES * 64 * 4);
  float* ro       = (float*)take((size_t)4 * N_NODES * 4);
  float* ri       = (float*)take((size_t)4 * N_NODES * 4);
  float* mbuf[4];
  for (int r = 0; r < 4; ++r) mbuf[r] = (float*)take((size_t)N_NODES * 64 * 4);
  float* acc_drug = (float*)take((size_t)N_NODES * 64 * 4);
  float* acc_dis  = (float*)take((size_t)N_NODES * 64 * 4);
  float* hb_drug  = (float*)take((size_t)N_NODES * 64 * 4);
  float* hb_dis   = (float*)take((size_t)N_NODES * 64 * 4);
  float* h2_drug  = (float*)take((size_t)N_NODES * 32 * 4);
  float* h2_dis   = (float*)take((size_t)N_NODES * 32 * 4);
  float* Pbuf     = (float*)take((size_t)N_NODES * 32 * 4);

  const int B = 256;
  // ---- zero accumulators
  k_zero<<<(N_NODES * 64 + B - 1) / B, B, 0, stream>>>(h_drug, N_NODES * 64);
  k_zero<<<(N_NODES * 64 + B - 1) / B, B, 0, stream>>>(h_dis,  N_NODES * 64);
  k_zero<<<(4 * N_NODES + B - 1) / B, B, 0, stream>>>(ro, 4 * N_NODES);
  k_zero<<<(4 * N_NODES + B - 1) / B, B, 0, stream>>>(ri, 4 * N_NODES);
  k_zero<<<(N_NODES * 64 + B - 1) / B, B, 0, stream>>>(acc_drug, N_NODES * 64);
  k_zero<<<(N_NODES * 64 + B - 1) / B, B, 0, stream>>>(acc_dis,  N_NODES * 64);

  // ---- weight swizzle + input embeddings (WMMA)
  k_swizzle_w<<<(KT * 4 * 32 * 8) / B, B, 0, stream>>>(W_drug, wbf_drug);
  k_swizzle_w<<<(KT * 4 * 32 * 8) / B, B, 0, stream>>>(W_dis,  wbf_dis);
  k_embed_gemm<<<dim3(125, NKCH), 128, 0, stream>>>(feat_drug, wbf_drug, h_drug);
  k_embed_gemm<<<dim3(125, NKCH), 128, 0, stream>>>(feat_dis,  wbf_dis,  h_dis);

  // ---- degree norms
  k_degrees<<<(4 * EDGES) / B, B, 0, stream>>>(src, dst, ro, ri);
  k_rsqrt_clip<<<(4 * N_NODES + B - 1) / B, B, 0, stream>>>(ro, 4 * N_NODES);
  k_rsqrt_clip<<<(4 * N_NODES + B - 1) / B, B, 0, stream>>>(ri, 4 * N_NODES);

  const float* hsrc1[4] = {h_drug, h_drug, h_dis, h_dis};
  // ---- hetero layer 1 (64 -> 64)
  for (int r = 0; r < 4; ++r)
    k_rel_gemm<<<(N_NODES * 64) / B, B, 0, stream>>>(hsrc1[r], ro + r * N_NODES,
                                                     W1 + (size_t)r * 64 * 64, mbuf[r], 6);
  for (int r = 0; r < 4; ++r) {
    float* acc = (r == 0 || r == 2) ? acc_drug : acc_dis;
    k_scatter<<<(EDGES * 16) / B, B, 0, stream>>>(mbuf[r], src + (size_t)r * EDGES,
                                                  dst + (size_t)r * EDGES,
                                                  ri + r * N_NODES, acc, 4, 6);
  }
  k_bias_relu<<<(N_NODES * 64) / B, B, 0, stream>>>(acc_drug, b1 + 0,  b1 + 128, hb_drug, 6);
  k_bias_relu<<<(N_NODES * 64) / B, B, 0, stream>>>(acc_dis,  b1 + 64, b1 + 192, hb_dis,  6);

  // ---- hetero layer 2 (64 -> 32)
  k_zero<<<(N_NODES * 32 + B - 1) / B, B, 0, stream>>>(acc_drug, N_NODES * 32);
  k_zero<<<(N_NODES * 32 + B - 1) / B, B, 0, stream>>>(acc_dis,  N_NODES * 32);
  const float* hsrc2[4] = {hb_drug, hb_drug, hb_dis, hb_dis};
  for (int r = 0; r < 4; ++r)
    k_rel_gemm<<<(N_NODES * 32) / B, B, 0, stream>>>(hsrc2[r], ro + r * N_NODES,
                                                     W2 + (size_t)r * 64 * 32, mbuf[r], 5);
  for (int r = 0; r < 4; ++r) {
    float* acc = (r == 0 || r == 2) ? acc_drug : acc_dis;
    k_scatter<<<(EDGES * 8) / B, B, 0, stream>>>(mbuf[r], src + (size_t)r * EDGES,
                                                 dst + (size_t)r * EDGES,
                                                 ri + r * N_NODES, acc, 3, 5);
  }
  k_bias_relu<<<(N_NODES * 32) / B, B, 0, stream>>>(acc_drug, b2 + 0,  b2 + 64, h2_drug, 5);
  k_bias_relu<<<(N_NODES * 32) / B, B, 0, stream>>>(acc_dis,  b2 + 32, b2 + 96, h2_dis,  5);

  // ---- bilinear scoring (WMMA, K = 32)
  k_bilinear_left<<<(N_NODES * 32) / B, B, 0, stream>>>(h2_drug, Wf, Pbuf);
  k_score_gemm<<<dim3(500, 125), 128, 0, stream>>>(Pbuf, h2_dis, out);
}
